// ViT_31164282700399
// MI455X (gfx1250) — matrix operand order repaired and checked by reading, not compiled, not run on hardware
//
#include <hip/hip_runtime.h>
#include <math.h>

#define B_     32
#define N_     257
#define D_     768
#define H_     12
#define DEPTH_ 12
#define HID_   3072
#define NCLS_  1000
#define HD_    64
#define SPAD_  288   // score row length padded to multiple of 32 (257 -> 288)

typedef unsigned short u16;
typedef __attribute__((ext_vector_type(16))) __bf16  v16bf;
typedef __attribute__((ext_vector_type(8)))  float   v8f;
typedef __attribute__((ext_vector_type(8)))  u16     us8;
typedef __attribute__((ext_vector_type(16))) u16     us16;
typedef __attribute__((ext_vector_type(4)))  u16     us4;
typedef __attribute__((ext_vector_type(4)))  float   f4;

__device__ __forceinline__ u16 f2bf(float f) {
  unsigned int u = __builtin_bit_cast(unsigned int, f);
  u += 0x7FFFu + ((u >> 16) & 1u);     // round-to-nearest-even
  return (u16)(u >> 16);
}

union FragU {
  struct { us8 lo; us8 hi; } h;
  us16  u;
  v16bf b;
};

// ---------------------------------------------------------------------------
// Batched GEMM: C[z] = act( A[z] (MxK) * B[z] + bias ) (+ Res)
// ABF/BBF : A / B stored as bf16 (u16) in global, else fp32 (converted in-loop)
// CBF     : C written as bf16 (u16), else fp32 (residual only in fp32 mode)
// TRANSB=true : B is Nn x K row-major; TRANSB=false: B is K x Nn row-major
// Launcher guarantees: K%32==0, lda%4==0, ldb%4==0, (TRANSB=false -> Nn%4==0),
// A rows have K readable columns (zero-padded), B rows >= Kb read as zero.
// flags: 1 = +bias[col], 2 = exact GELU, 4 = += Res[ci] (fp32 C only)
// batch z -> (bb = z / Hdiv, hh = z % Hdiv)
// ---------------------------------------------------------------------------
template<bool ABF, bool BBF, bool TRANSB, bool CBF>
__global__ __launch_bounds__(256) void gemm_wmma_bf16(
    const void* __restrict__ Av, const void* __restrict__ Bv,
    void* __restrict__ Cv, const float* __restrict__ bias,
    const float* __restrict__ Res,
    int M, int Nn, int K, int Kb, int lda, int ldb, int ldc,
    long sAb, long sAh, long sBb, long sBh, long sCb, long sCh,
    int Hdiv, int flags)
{
  __shared__ u16 As[2][128 * 32];
  __shared__ u16 Bs[2][128 * 32];

  const float* Af = (const float*)Av;
  const u16*   Ah = (const u16*)Av;
  const float* Bf = (const float*)Bv;
  const u16*   Bh = (const u16*)Bv;
  float*       Cf = (float*)Cv;
  u16*         Ch = (u16*)Cv;

  const int z  = blockIdx.z;
  const int bb = z / Hdiv, hh = z % Hdiv;
  const long aOff = (long)bb * sAb + (long)hh * sAh;
  const long bOff = (long)bb * sBb + (long)hh * sBh;
  const long cOff = (long)bb * sCb + (long)hh * sCh;

  const int m0   = blockIdx.x * 128;
  const int n0   = blockIdx.y * 128;
  const int tid  = threadIdx.x;
  const int wave = tid >> 5, lane = tid & 31;
  const int wm   = wave & 3, wn = wave >> 2;     // 4x2 wave grid: 32x64 C per wave
  const int lr   = lane & 15, half = lane >> 4;

  v8f acc[2][4];
  const v8f vzero = {0.f, 0.f, 0.f, 0.f, 0.f, 0.f, 0.f, 0.f};
#pragma unroll
  for (int mi = 0; mi < 2; ++mi)
#pragma unroll
    for (int ni = 0; ni < 4; ++ni) acc[mi][ni] = vzero;

  // per-thread tile coordinates for staged loads
  const int ra = tid >> 3;             // row base (A and TRANSB-B), +32 per it
  const int ca = (tid & 7) << 2;       // k-column base
  const int kr = tid >> 5;             // k-row (B transpose path), +8 per it
  const int nc = (tid & 31) << 2;      // n-column (B transpose path)

  f4  va[4], vb[4];
  us4 ua[4], ub[4];
  const us4 uzero = {0, 0, 0, 0};
  const f4  fzero = {0.f, 0.f, 0.f, 0.f};

  auto loadA = [&](int k0) {
#pragma unroll
    for (int it = 0; it < 4; ++it) {
      int gm = m0 + ra + it * 32;
      if constexpr (ABF) {
        us4 v = uzero;
        if (gm < M) v = *(const us4*)(Ah + aOff + (long)gm * lda + (k0 + ca));
        ua[it] = v;
      } else {
        f4 v = fzero;
        if (gm < M) v = *(const f4*)(Af + aOff + (long)gm * lda + (k0 + ca));
        va[it] = v;
      }
    }
  };
  auto storeA = [&](int buf) {
#pragma unroll
    for (int it = 0; it < 4; ++it) {
      int r = ra + it * 32;
      us4 p;
      if constexpr (ABF) {
        p = ua[it];
      } else {
#pragma unroll
        for (int e = 0; e < 4; ++e) p[e] = f2bf(va[it][e]);
      }
      *(us4*)&As[buf][r * 32 + ca] = p;
    }
  };
  auto loadB = [&](int k0) {
    if constexpr (TRANSB) {
#pragma unroll
      for (int it = 0; it < 4; ++it) {
        int gn = n0 + ra + it * 32;
        if constexpr (BBF) {
          us4 v = uzero;
          if (gn < Nn) v = *(const us4*)(Bh + bOff + (long)gn * ldb + (k0 + ca));
          ub[it] = v;
        } else {
          f4 v = fzero;
          if (gn < Nn) v = *(const f4*)(Bf + bOff + (long)gn * ldb + (k0 + ca));
          vb[it] = v;
        }
      }
    } else {
#pragma unroll
      for (int it = 0; it < 4; ++it) {
        int gk = k0 + kr + it * 8;
        bool ok = (gk < Kb) && ((n0 + nc) < Nn);
        if constexpr (BBF) {
          us4 v = uzero;
          if (ok) v = *(const us4*)(Bh + bOff + (long)gk * ldb + (n0 + nc));
          ub[it] = v;
        } else {
          f4 v = fzero;
          if (ok) v = *(const f4*)(Bf + bOff + (long)gk * ldb + (n0 + nc));
          vb[it] = v;
        }
      }
    }
  };
  auto storeB = [&](int buf) {
    if constexpr (TRANSB) {
#pragma unroll
      for (int it = 0; it < 4; ++it) {
        int r = ra + it * 32;
        us4 p;
        if constexpr (BBF) {
          p = ub[it];
        } else {
#pragma unroll
          for (int e = 0; e < 4; ++e) p[e] = f2bf(vb[it][e]);
        }
        *(us4*)&Bs[buf][r * 32 + ca] = p;
      }
    } else {
#pragma unroll
      for (int it = 0; it < 4; ++it) {
        int k = kr + it * 8;
#pragma unroll
        for (int e = 0; e < 4; ++e) {
          u16 p;
          if constexpr (BBF) p = ub[it][e]; else p = f2bf(vb[it][e]);
          Bs[buf][(nc + e) * 32 + k] = p;
        }
      }
    }
  };

  // ---- software pipeline ----
  loadA(0); loadB(0);
  storeA(0); storeB(0);
  __syncthreads();

  int buf = 0;
  for (int k0 = 0; k0 < K; k0 += 32) {
    const int kn = k0 + 32;
    if (kn < K) { loadA(kn); loadB(kn); }

    // prefetch tile t+2 (global_prefetch_b8)
    const int kp = k0 + 64;
    if (kp < K) {
      int gm = m0 + ra;
      if (gm < M) {
        if constexpr (ABF) __builtin_prefetch(Ah + aOff + (long)gm * lda + (kp + ca), 0, 1);
        else               __builtin_prefetch(Af + aOff + (long)gm * lda + (kp + ca), 0, 1);
      }
    }

    // ---- compute on buffer `buf` ----
    const u16* Ab = As[buf];
    const u16* Bb = Bs[buf];
    FragU a[2];
#pragma unroll
    for (int mi = 0; mi < 2; ++mi) {
      int m = wm * 32 + mi * 16 + lr;
      a[mi].h.lo = *(const us8*)&Ab[m * 32 + half * 16];      // K = 16h + 0..7
      a[mi].h.hi = *(const us8*)&Ab[m * 32 + half * 16 + 8];  // K = 16h + 8..15
    }
#pragma unroll
    for (int ni = 0; ni < 4; ++ni) {
      int n = wn * 64 + ni * 16 + lr;
      FragU bf;
      bf.h.lo = *(const us8*)&Bb[n * 32 + half * 16];         // K = 16h + 0..7
      bf.h.hi = *(const us8*)&Bb[n * 32 + half * 16 + 8];     // K = 16h + 8..15
#pragma unroll
      for (int mi = 0; mi < 2; ++mi)
        acc[mi][ni] = __builtin_amdgcn_wmma_f32_16x16x32_bf16(
            false, a[mi].b, false, bf.b, (short)0, acc[mi][ni], false, false);
    }

    if (kn < K) { storeA(buf ^ 1); storeB(buf ^ 1); }
    __syncthreads();
    buf ^= 1;
  }

  // ---- epilogue: bias / GELU / residual, guarded stores ----
#pragma unroll
  for (int mi = 0; mi < 2; ++mi) {
#pragma unroll
    for (int ni = 0; ni < 4; ++ni) {
      int col = n0 + wn * 64 + ni * 16 + lr;
      if (col >= Nn) continue;
      float bv = (flags & 1) ? bias[col] : 0.f;
#pragma unroll
      for (int r = 0; r < 8; ++r) {
        int row = m0 + wm * 32 + mi * 16 + half * 8 + r;
        if (row >= M) continue;
        float v = acc[mi][ni][r] + bv;
        if (flags & 2) v = 0.5f * v * (1.f + erff(v * 0.70710678118f));
        long ci = cOff + (long)row * ldc + col;
        if constexpr (CBF) {
          Ch[ci] = f2bf(v);
        } else {
          if (flags & 4) v += Res[ci];
          Cf[ci] = v;
        }
      }
    }
  }
}

// ---------------------------------------------------------------------------
// LayerNorm: fp32 in, bf16 out (feeds GEMM A-side)
__global__ __launch_bounds__(256) void layernorm_bf16_k(
    const float* __restrict__ x, const float* __restrict__ g,
    const float* __restrict__ b, u16* __restrict__ o, int d)
{
  __shared__ float s1[256], s2[256];
  long row = blockIdx.x;
  const float* xr = x + row * d;
  u16* orow = o + row * d;
  int tid = threadIdx.x;
  float a1 = 0.f, a2 = 0.f;
  for (int i = tid; i < d; i += 256) { float v = xr[i]; a1 += v; a2 += v * v; }
  s1[tid] = a1; s2[tid] = a2; __syncthreads();
  for (int off = 128; off; off >>= 1) {
    if (tid < off) { s1[tid] += s1[tid + off]; s2[tid] += s2[tid + off]; }
    __syncthreads();
  }
  float mu  = s1[0] / d;
  float var = s2[0] / d - mu * mu;
  float rstd = rsqrtf(var + 1e-5f);
  for (int i = tid; i < d; i += 256)
    orow[i] = f2bf((xr[i] - mu) * rstd * g[i] + b[i]);
}

// softmax rows: n valid, padded to npad; fuses (s + pos)*scale; zeroes pad cols.
__global__ __launch_bounds__(256) void softmax_k(
    float* __restrict__ sc, const float* __restrict__ pos,
    int n, int npad, float scale)
{
  __shared__ float red[256];
  long row = blockIdx.x;
  float* sr = sc + row * (long)npad;
  const float* pr = pos + row * (long)n;
  int tid = threadIdx.x;
  float mx = -3.4e38f;
  for (int i = tid; i < n; i += 256) {
    float v = (sr[i] + pr[i]) * scale;
    sr[i] = v;
    mx = fmaxf(mx, v);
  }
  red[tid] = mx; __syncthreads();
  for (int off = 128; off; off >>= 1) {
    if (tid < off) red[tid] = fmaxf(red[tid], red[tid + off]);
    __syncthreads();
  }
  mx = red[0]; __syncthreads();
  float s = 0.f;
  for (int i = tid; i < n; i += 256) { float e = expf(sr[i] - mx); sr[i] = e; s += e; }
  red[tid] = s; __syncthreads();
  for (int off = 128; off; off >>= 1) {
    if (tid < off) red[tid] += red[tid + off];
    __syncthreads();
  }
  float inv = 1.f / red[0];
  for (int i = tid; i < n; i += 256) sr[i] *= inv;
  for (int i = n + tid; i < npad; i += 256) sr[i] = 0.f;
}

// ---------------------------------------------------------------------------
extern "C" void kernel_launch(void* const* d_in, const int* in_sizes, int n_in,
                              void* d_out, int out_size, void* d_ws, size_t ws_size,
                              hipStream_t stream)
{
  (void)in_sizes; (void)n_in; (void)out_size; (void)ws_size;
  const float* x_in   = (const float*)d_in[0];
  const float* pos    = (const float*)d_in[1];
  const float* qkv_w  = (const float*)d_in[2];
  const float* qkv_b  = (const float*)d_in[3];
  const float* proj_w = (const float*)d_in[4];
  const float* proj_b = (const float*)d_in[5];
  const float* ln1_g  = (const float*)d_in[6];
  const float* ln1_b  = (const float*)d_in[7];
  const float* ln2_g  = (const float*)d_in[8];
  const float* ln2_b  = (const float*)d_in[9];
  const float* fc1_w  = (const float*)d_in[10];
  const float* fc1_b  = (const float*)d_in[11];
  const float* fc2_w  = (const float*)d_in[12];
  const float* fc2_b  = (const float*)d_in[13];
  const float* lnf_g  = (const float*)d_in[14];
  const float* lnf_b  = (const float*)d_in[15];
  const float* head_w = (const float*)d_in[16];
  const float* head_b = (const float*)d_in[17];
  float* out = (float*)d_out;

  const long BND = (long)B_ * N_ * D_;              // 6,316,032
  const long QKV = (long)B_ * N_ * 3 * D_;          // 18,948,096
  float* ws   = (float*)d_ws;
  float* xw   = ws;                                  // fp32 residual stream
  u16*   hb   = (u16*)(xw + BND);                    // bf16 LN output
  u16*   qkvb = hb + BND;                            // bf16 fused qkv
  u16*   aob  = qkvb + QKV;                          // bf16 attention output
  float* sh   = (float*)(aob + BND);                 // fp32 padded scores / bf16 mlp hidden

  hipMemcpyAsync(xw, x_in, BND * sizeof(float), hipMemcpyDeviceToDevice, stream);

  const int Mrows = B_ * N_;                         // 8224
  const long zero = 0;

  for (int i = 0; i < DEPTH_; ++i) {
    // ---- attention ----
    layernorm_bf16_k<<<Mrows, 256, 0, stream>>>(xw, ln1_g + (long)i * D_, ln1_b + (long)i * D_, hb, D_);

    // qkv = h @ Wqkv + b   (A bf16, B fp32 weights, C bf16)
    gemm_wmma_bf16<true, false, false, true><<<dim3(65, 18, 1), 256, 0, stream>>>(
        hb, qkv_w + (long)i * D_ * 3 * D_, qkvb, qkv_b + (long)i * 3 * D_, nullptr,
        Mrows, 3 * D_, D_, D_, D_, 3 * D_, 3 * D_,
        zero, zero, zero, zero, zero, zero, 1, /*bias*/1);

    // scores = q . k^T  (both bf16, no in-loop conversion), fp32 padded output
    gemm_wmma_bf16<true, true, true, false><<<dim3(3, 3, B_ * H_), 256, 0, stream>>>(
        qkvb /*q*/, qkvb + D_ /*k*/, sh, nullptr, nullptr,
        N_, N_, HD_, HD_, 3 * D_, 3 * D_, SPAD_,
        (long)N_ * 3 * D_, (long)HD_, (long)N_ * 3 * D_, (long)HD_,
        (long)H_ * N_ * SPAD_, (long)N_ * SPAD_, H_, 0);

    softmax_k<<<B_ * H_ * N_, 256, 0, stream>>>(sh, pos, N_, SPAD_, 0.125f);

    // out = attn . v   (A fp32 attn, B bf16 v, C bf16) -> aob as [B, N, H*HD]
    gemm_wmma_bf16<false, true, false, true><<<dim3(3, 1, B_ * H_), 256, 0, stream>>>(
        sh, qkvb + 2 * D_ /*v*/, aob, nullptr, nullptr,
        N_, HD_, SPAD_, N_, SPAD_, 3 * D_, D_,
        (long)H_ * N_ * SPAD_, (long)N_ * SPAD_, (long)N_ * 3 * D_, (long)HD_,
        (long)N_ * D_, (long)HD_, H_, 0);

    // x += out @ Wproj + b  (A bf16, C fp32 + residual)
    gemm_wmma_bf16<true, false, false, false><<<dim3(65, 6, 1), 256, 0, stream>>>(
        aob, proj_w + (long)i * D_ * D_, xw, proj_b + (long)i * D_, xw,
        Mrows, D_, D_, D_, D_, D_, D_,
        zero, zero, zero, zero, zero, zero, 1, /*bias|res*/ 1 | 4);

    // ---- mlp ----
    layernorm_bf16_k<<<Mrows, 256, 0, stream>>>(xw, ln2_g + (long)i * D_, ln2_b + (long)i * D_, hb, D_);

    // hid = gelu(h @ W1 + b)  (C bf16 into sh)
    gemm_wmma_bf16<true, false, false, true><<<dim3(65, 24, 1), 256, 0, stream>>>(
        hb, fc1_w + (long)i * D_ * HID_, (u16*)sh, fc1_b + (long)i * HID_, nullptr,
        Mrows, HID_, D_, D_, D_, HID_, HID_,
        zero, zero, zero, zero, zero, zero, 1, /*bias|gelu*/ 1 | 2);

    // x += hid @ W2 + b
    gemm_wmma_bf16<true, false, false, false><<<dim3(65, 6, 1), 256, 0, stream>>>(
        (u16*)sh, fc2_w + (long)i * HID_ * D_, xw, fc2_b + (long)i * D_, xw,
        Mrows, D_, HID_, HID_, HID_, D_, D_,
        zero, zero, zero, zero, zero, zero, 1, /*bias|res*/ 1 | 4);
  }

  // ---- final LN + classifier head on CLS tokens (strided bf16 rows) ----
  layernorm_bf16_k<<<Mrows, 256, 0, stream>>>(xw, lnf_g, lnf_b, hb, D_);

  gemm_wmma_bf16<true, false, false, false><<<dim3(1, 8, 1), 256, 0, stream>>>(
      hb, head_w, out, head_b, nullptr,
      B_, NCLS_, D_, D_, N_ * D_ /*lda: CLS->CLS*/, NCLS_, NCLS_,
      zero, zero, zero, zero, zero, zero, 1, /*bias*/1);
}
